// NeuralMemory_6012954214664
// MI455X (gfx1250) — matrix operand order repaired and checked by reading, not compiled, not run on hardware
//
#include <hip/hip_runtime.h>

// Problem constants from the reference
constexpr int B = 4, L = 2048, D = 512, H = 512;
constexpr int LT = L / 16;          // 128 l-tiles per batch

typedef __attribute__((ext_vector_type(16))) __bf16   v16bf;
typedef __attribute__((ext_vector_type(8)))  float    v8f;
typedef __attribute__((ext_vector_type(4)))  unsigned int uint32x4;
// TDM descriptor group types (6-arg builtin signature: v4u, v8i, v4i, v4i, v8i, i32)
typedef __attribute__((ext_vector_type(4))) unsigned int tdm_g0_t;
typedef __attribute__((ext_vector_type(8))) int          tdm_g1_t;
typedef __attribute__((ext_vector_type(4))) int          tdm_g4_t;

union FragAB { v16bf v; uint32x4 q[2]; };

__device__ __forceinline__ unsigned short f2bf(float f) {
    unsigned int u = __float_as_uint(f);
    unsigned int r = u + 0x7FFFu + ((u >> 16) & 1u);   // round-to-nearest-even
    return (unsigned short)(r >> 16);
}

// ---------------------------------------------------------------- casts
__global__ void cast_bf16_kernel(const float* __restrict__ src,
                                 unsigned short* __restrict__ dst, size_t n) {
    for (size_t i = (size_t)blockIdx.x * blockDim.x + threadIdx.x; i < n;
         i += (size_t)gridDim.x * blockDim.x)
        dst[i] = f2bf(src[i]);
}

// dst[k*N + n] = bf16(src[n*K + k])   (src is (N,K) row-major)
__global__ void transpose_bf16_kernel(const float* __restrict__ src,
                                      unsigned short* __restrict__ dst,
                                      int N, int K) {
    size_t total = (size_t)N * K;
    for (size_t i = (size_t)blockIdx.x * blockDim.x + threadIdx.x; i < total;
         i += (size_t)gridDim.x * blockDim.x) {
        int k = (int)(i / N), n = (int)(i % N);
        dst[i] = f2bf(src[(size_t)n * K + k]);
    }
}

// hT[b][h][d] = bf16(hidden[b][d][h])
__global__ void hidden_t_kernel(const float* __restrict__ hidden,
                                unsigned short* __restrict__ hT) {
    size_t total = (size_t)B * H * D;
    for (size_t i = (size_t)blockIdx.x * blockDim.x + threadIdx.x; i < total;
         i += (size_t)gridDim.x * blockDim.x) {
        int b = (int)(i / ((size_t)H * D));
        int rem = (int)(i % ((size_t)H * D));
        int h = rem / D, d = rem % D;
        hT[i] = f2bf(hidden[((size_t)b * D + d) * H + h]);
    }
}

// ---------------------------------------------------------------- per-token scalars
__global__ void token_scalars_kernel(const float* __restrict__ x,
                                     const float* __restrict__ flw,
                                     const float* __restrict__ flb,
                                     const float* __restrict__ fww,
                                     const float* __restrict__ fwb,
                                     float* __restrict__ zlr,
                                     float* __restrict__ zwd) {
    const int lane = threadIdx.x & 31;
    const int r = (int)((blockIdx.x * (size_t)blockDim.x + threadIdx.x) >> 5);
    const float* xr = x + (size_t)r * D;
    float a = 0.f, c = 0.f;
    for (int i = lane; i < D; i += 32) {
        float v = xr[i];
        a += v * flw[i];
        c += v * fww[i];
    }
    for (int m = 16; m >= 1; m >>= 1) {
        a += __shfl_xor(a, m, 32);
        c += __shfl_xor(c, m, 32);
    }
    if (lane == 0) { zlr[r] = a + flb[0]; zwd[r] = c + fwb[0]; }
}

// ---------------------------------------------------------------- per-batch prefix scan
__global__ void scan_kernel(const float* __restrict__ zlr,
                            const float* __restrict__ zwd,
                            const float* __restrict__ lbl_p,
                            const float* __restrict__ lbw_p,
                            float* __restrict__ lr, float* __restrict__ C,
                            float* __restrict__ wdc, float* __restrict__ coef,
                            float* __restrict__ wdlast) {
    __shared__ float s[257];
    const int b = blockIdx.x, t = threadIdx.x;
    const float lbl = lbl_p[0], lbw = lbw_p[0];
    const int base = b * L + t * 8;
    float vals[8];
    float run = 0.f;
#pragma unroll
    for (int i = 0; i < 8; ++i) {
        float z = zwd[base + i];
        float ls = (z >= 0.f) ? -log1pf(__expf(-z)) : (z - log1pf(__expf(z)));
        run += lbw + ls;
        vals[i] = run;
    }
    s[t] = run;
    __syncthreads();
    if (t == 0) {
        float acc = 0.f;
        for (int i = 0; i < 256; ++i) { float v = s[i]; s[i] = acc; acc += v; }
        s[256] = acc;                       // C[L-1]
    }
    __syncthreads();
    const float excl = s[t], Ctot = s[256];
    const float elbl = __expf(lbl);
#pragma unroll
    for (int i = 0; i < 8; ++i) {
        int idx = base + i;
        float Ci = excl + vals[i];
        C[idx]   = Ci;
        wdc[idx] = __expf(Ci);
        float lri = elbl / (1.f + __expf(-zlr[idx]));
        lr[idx]  = lri;
        coef[idx] = lri * __expf(Ctot - Ci);
    }
    if (t == 0) wdlast[b] = __expf(Ctot);
}

// ---------------------------------------------------------------- GEMM: out = Xbf (R x K) * WT (K x N)
// Block = 8 r-tiles x 1 n-tile. B panel (512x16 bf16, 16KB) staged in LDS by the
// Tensor Data Mover (one DMA per block), A streamed from global into WMMA.
// mode 0: bf16 transposed per batch (Qt[b][n][l]);  1: bf16 row-major;  2: f32 row-major
__global__ void gemm_xw_kernel(const unsigned short* __restrict__ Xbf,
                               const unsigned short* __restrict__ WT,
                               float* __restrict__ outF,
                               unsigned short* __restrict__ outB16,
                               unsigned short* __restrict__ outB16T,
                               int mode) {
    __shared__ __align__(16) unsigned short bpanel[D * 16];   // [k][16]
    const int lane = threadIdx.x & 31;
    const int wave = threadIdx.x >> 5;
    const int nblk = (int)blockIdx.x & 31;          // 32 n-tiles
    const int rblk = (int)blockIdx.x >> 5;          // 64 r-blocks of 128 rows
    const int n0 = nblk * 16;
    const int r0 = rblk * 128 + wave * 16;

    if (threadIdx.x < 32) {
        // TDM descriptor: 2-D tensor WT (512 rows x 512 cols, bf16),
        // tile = 16 cols x 512 rows starting at column n0 -> contiguous 16KB in LDS.
        unsigned long long ga = (unsigned long long)(size_t)(WT + n0);
        unsigned int lds = (unsigned int)(size_t)(&bpanel[0]);
        tdm_g0_t g0 = { 1u,                                  // count=1, user mode
                        lds,                                 // lds_addr
                        (unsigned int)ga,                    // global_addr[31:0]
                        (unsigned int)((ga >> 32) & 0x01FFFFFFu) | (2u << 30) }; // [56:32] + type=2
        tdm_g1_t g1 = { (int)(1u << 16),      // data_size=1 (2 bytes); wg_mask=0
                        (int)((unsigned)D << 16),            // tensor_dim0[15:0] << 16
                        (int)((unsigned)D << 16),            // dim0 hi=0 | tensor_dim1 lo << 16
                        (int)(16u << 16),                    // dim1 hi=0 | tile_dim0=16
                        (int)D,                              // tile_dim1=512 | tile_dim2=0
                        (int)D,                              // tensor_dim0_stride lo
                        0, 0 };                              // stride0 hi | stride1
        tdm_g4_t gz4 = { 0, 0, 0, 0 };
        tdm_g1_t gz8 = { 0, 0, 0, 0, 0, 0, 0, 0 };
        __builtin_amdgcn_tensor_load_to_lds(g0, g1, gz4, gz4, gz8, 0);
        __builtin_amdgcn_s_wait_tensorcnt(0);
    }
    __syncthreads();

    const int arow = r0 + (lane & 15);
    const int koff = (lane & 16) ? 8 : 0;
    v8f acc = {};
    for (int k0 = 0; k0 < D; k0 += 32) {
        FragAB a, bm;
        const unsigned short* pa = Xbf + (size_t)arow * D + k0 + koff;
        a.q[0] = *(const uint32x4*)(pa);
        a.q[1] = *(const uint32x4*)(pa + 16);
        for (int i = 0; i < 8; ++i) {
            int kk = k0 + koff + (i & 3) * 2 + (i >> 2) * 16;
            bm.q[i >> 2][i & 3] = (unsigned)bpanel[kk * 16 + (lane & 15)] | ((unsigned)bpanel[(kk + 1) * 16 + (lane & 15)] << 16);
        }
        acc = __builtin_amdgcn_wmma_f32_16x16x32_bf16(false, a.v, false, bm.v,
                                                      (short)0, acc, false, false);
    }
    const int roff = (lane & 16) ? 8 : 0;
    const int col = n0 + (lane & 15);
#pragma unroll
    for (int j = 0; j < 8; ++j) {
        int r = r0 + j + roff;
        float v = acc[j];
        if (mode == 0) {
            int bb = r >> 11, l = r & (L - 1);
            outB16T[((size_t)(bb * H + col)) * L + l] = f2bf(v);
        } else if (mode == 1) {
            outB16[(size_t)r * H + col] = f2bf(v);
        } else {
            outF[(size_t)r * H + col] = v;
        }
    }
}

// ---------------------------------------------------------------- WpK = K * hT; emits y_cross, WpK-V, Ut
__global__ void gemm_wpk_kernel(const unsigned short* __restrict__ Kbf,
                                const unsigned short* __restrict__ hT,
                                const float* __restrict__ V,
                                const float* __restrict__ wdc,
                                const float* __restrict__ coef,
                                float* __restrict__ y_out,
                                float* __restrict__ WpKV,
                                unsigned short* __restrict__ Ut) {
    const int lane = threadIdx.x & 31;
    const int wid = (int)((blockIdx.x * (size_t)blockDim.x + threadIdx.x) >> 5);
    const int tiles_n = D / 16;
    const int r0 = (wid / tiles_n) * 16;
    const int n0 = (wid % tiles_n) * 16;
    const int b = r0 >> 11;
    const int arow = r0 + (lane & 15);
    const int koff = (lane & 16) ? 8 : 0;
    const unsigned short* hTb = hT + (size_t)b * H * D;
    v8f acc = {};
    for (int k0 = 0; k0 < H; k0 += 32) {
        FragAB a, bm;
        const unsigned short* pa = Kbf + (size_t)arow * H + k0 + koff;
        a.q[0] = *(const uint32x4*)(pa);
        a.q[1] = *(const uint32x4*)(pa + 16);
        const unsigned short* pb = hTb + n0 + (lane & 15);
        for (int i = 0; i < 8; ++i) {
            int kk = k0 + koff + (i & 3) * 2 + (i >> 2) * 16;
            bm.q[i >> 2][i & 3] = (unsigned)pb[(size_t)kk * D] | ((unsigned)pb[(size_t)(kk + 1) * D] << 16);
        }
        acc = __builtin_amdgcn_wmma_f32_16x16x32_bf16(false, a.v, false, bm.v,
                                                      (short)0, acc, false, false);
    }
    const int roff = (lane & 16) ? 8 : 0;
    const int col = n0 + (lane & 15);
#pragma unroll
    for (int j = 0; j < 8; ++j) {
        int r = r0 + j + roff;
        size_t idx = (size_t)r * D + col;
        float wpk = acc[j];
        float wpkv = wpk - V[idx];
        WpKV[idx] = wpkv;
        y_out[idx] = wpk * wdc[r];                    // y_cross contribution
        int l = r & (L - 1);
        Ut[((size_t)(b * D + col)) * L + l] = f2bf(-wpkv * coef[r]);
    }
}

// ---------------------------------------------------------------- S[l] = sum_{m>=l} (K[l].Q[m]) exp(C[m]-C[l])
__global__ void kq_rowsum_kernel(const unsigned short* __restrict__ Kbf,
                                 const unsigned short* __restrict__ Qt,
                                 const float* __restrict__ C,
                                 float* __restrict__ S) {
    const int lane = threadIdx.x & 31;
    const int wid = (int)((blockIdx.x * (size_t)blockDim.x + threadIdx.x) >> 5);
    const int b = wid >> 7;
    const int lt = wid & (LT - 1);
    const int l0 = lt * 16;
    const float* Cb = C + (size_t)b * L;
    const int roff = (lane & 16) ? 8 : 0;
    float cl[8];
#pragma unroll
    for (int j = 0; j < 8; ++j) cl[j] = Cb[l0 + j + roff];
    const float clmax = Cb[l0 + 15];
    const unsigned short* Kb = Kbf + (size_t)b * L * H;
    const unsigned short* Qb = Qt + (size_t)b * H * L;
    const int arow = l0 + (lane & 15);
    const int koff = (lane & 16) ? 8 : 0;
    float accS[8];
#pragma unroll
    for (int j = 0; j < 8; ++j) accS[j] = 0.f;

    for (int mt = lt; mt < LT; ++mt) {
        const int m0 = mt * 16;
        if (Cb[m0] - clmax < -90.f) break;    // decay underflow: all later tiles 0
        v8f kq = {};
        for (int h0 = 0; h0 < H; h0 += 32) {
            FragAB a, bq;
            const unsigned short* pa = Kb + (size_t)arow * H + h0 + koff;
            a.q[0] = *(const uint32x4*)(pa);
            a.q[1] = *(const uint32x4*)(pa + 16);
            const unsigned short* pb = Qb + m0 + (lane & 15);
            for (int i = 0; i < 8; ++i) {
                int kk = h0 + koff + (i & 3) * 2 + (i >> 2) * 16;
                bq.q[i >> 2][i & 3] = (unsigned)pb[(size_t)kk * L] | ((unsigned)pb[(size_t)(kk + 1) * L] << 16);
            }
            kq = __builtin_amdgcn_wmma_f32_16x16x32_bf16(false, a.v, false, bq.v,
                                                         (short)0, kq, false, false);
        }
        const int me = m0 + (lane & 15);
        const float cm = Cb[me];
#pragma unroll
        for (int j = 0; j < 8; ++j) {
            int le = l0 + j + roff;
            float w = (me >= le) ? __expf(cm - cl[j]) : 0.f;
            accS[j] += kq[j] * w;
        }
    }
#pragma unroll
    for (int j = 0; j < 8; ++j) {
        float v = accS[j];
        v += __shfl_xor(v, 1, 32);
        v += __shfl_xor(v, 2, 32);
        v += __shfl_xor(v, 4, 32);
        v += __shfl_xor(v, 8, 32);
        if ((lane & 15) == 0)
            S[(size_t)b * L + l0 + j + ((lane >> 4) << 3)] = v;
    }
}

// ---------------------------------------------------------------- y += -(WpK-V) * lr * S
__global__ void finalize_y_kernel(float* __restrict__ y,
                                  const float* __restrict__ WpKV,
                                  const float* __restrict__ lr,
                                  const float* __restrict__ S) {
    size_t total = (size_t)B * L * D;
    for (size_t i = (size_t)blockIdx.x * blockDim.x + threadIdx.x; i < total;
         i += (size_t)gridDim.x * blockDim.x) {
        size_t r = i >> 9;                           // / D
        y[i] = y[i] - WpKV[i] * lr[r] * S[r];
    }
}

// ---------------------------------------------------------------- W_next = Ut * K + hidden * wd_last
__global__ void gemm_wnext_kernel(const unsigned short* __restrict__ Ut,
                                  const unsigned short* __restrict__ Kbf,
                                  const float* __restrict__ hidden,
                                  const float* __restrict__ wdlast,
                                  float* __restrict__ Wn) {
    const int lane = threadIdx.x & 31;
    const int wid = (int)((blockIdx.x * (size_t)blockDim.x + threadIdx.x) >> 5);
    const int tiles_h = H / 16, tiles_d = D / 16;
    const int b = wid / (tiles_h * tiles_d);
    const int rem = wid % (tiles_h * tiles_d);
    const int d0 = (rem / tiles_h) * 16;
    const int h0 = (rem % tiles_h) * 16;
    const unsigned short* Ub = Ut + (size_t)b * D * L;
    const unsigned short* Kb = Kbf + (size_t)b * L * H;
    const int arow = d0 + (lane & 15);
    const int koff = (lane & 16) ? 8 : 0;
    v8f acc = {};
    for (int k0 = 0; k0 < L; k0 += 32) {
        FragAB a, bm;
        const unsigned short* pa = Ub + (size_t)arow * L + k0 + koff;
        a.q[0] = *(const uint32x4*)(pa);
        a.q[1] = *(const uint32x4*)(pa + 16);
        const unsigned short* pb = Kb + h0 + (lane & 15);
        for (int i = 0; i < 8; ++i) {
            int kk = k0 + koff + (i & 3) * 2 + (i >> 2) * 16;
            bm.q[i >> 2][i & 3] = (unsigned)pb[(size_t)kk * H] | ((unsigned)pb[(size_t)(kk + 1) * H] << 16);
        }
        acc = __builtin_amdgcn_wmma_f32_16x16x32_bf16(false, a.v, false, bm.v,
                                                      (short)0, acc, false, false);
    }
    const float wl = wdlast[b];
    const int roff = (lane & 16) ? 8 : 0;
    const int col = h0 + (lane & 15);
#pragma unroll
    for (int j = 0; j < 8; ++j) {
        int d = d0 + j + roff;
        size_t idx = (size_t)b * D * H + (size_t)d * H + col;
        Wn[idx] = acc[j] + hidden[idx] * wl;
    }
}

// ---------------------------------------------------------------- launcher
extern "C" void kernel_launch(void* const* d_in, const int* in_sizes, int n_in,
                              void* d_out, int out_size, void* d_ws, size_t ws_size,
                              hipStream_t stream) {
    const float* x      = (const float*)d_in[0];
    const float* hidden = (const float*)d_in[1];
    const float* lbl    = (const float*)d_in[2];
    const float* lbw    = (const float*)d_in[3];
    const float* Wq     = (const float*)d_in[4];
    const float* Wk     = (const float*)d_in[5];
    const float* Wv     = (const float*)d_in[6];
    const float* flw    = (const float*)d_in[7];
    const float* flb    = (const float*)d_in[8];
    const float* fww    = (const float*)d_in[9];
    const float* fwb    = (const float*)d_in[10];

    float* y  = (float*)d_out;                       // (B,L,D)
    float* Wn = (float*)d_out + (size_t)B * L * D;   // (B,D,H)

    const size_t NTOK = (size_t)B * L;
    const size_t XSZ = NTOK * D;

    char* w = (char*)d_ws;
    auto take = [&](size_t bytes) { char* p = w; w += (bytes + 255) & ~(size_t)255; return p; };
    unsigned short* Xbf  = (unsigned short*)take(XSZ * 2);
    unsigned short* WqT  = (unsigned short*)take((size_t)D * H * 2);
    unsigned short* WkT  = (unsigned short*)take((size_t)D * H * 2);
    unsigned short* WvT  = (unsigned short*)take((size_t)D * D * 2);
    unsigned short* hT   = (unsigned short*)take((size_t)B * H * D * 2);
    unsigned short* Qt   = (unsigned short*)take((size_t)B * H * L * 2);
    unsigned short* Kbf  = (unsigned short*)take(XSZ * 2);
    float* V     = (float*)take(XSZ * 4);
    float* WpKV  = (float*)take(XSZ * 4);
    unsigned short* Ut = (unsigned short*)take((size_t)B * D * L * 2);
    float* zlr   = (float*)take(NTOK * 4);
    float* zwd   = (float*)take(NTOK * 4);
    float* lr    = (float*)take(NTOK * 4);
    float* C     = (float*)take(NTOK * 4);
    float* wdc   = (float*)take(NTOK * 4);
    float* coef  = (float*)take(NTOK * 4);
    float* S     = (float*)take(NTOK * 4);
    float* wdlast = (float*)take((size_t)B * 4);

    // casts / transposes
    cast_bf16_kernel<<<4096, 256, 0, stream>>>(x, Xbf, XSZ);
    transpose_bf16_kernel<<<1024, 256, 0, stream>>>(Wq, WqT, H, D);
    transpose_bf16_kernel<<<1024, 256, 0, stream>>>(Wk, WkT, H, D);
    transpose_bf16_kernel<<<1024, 256, 0, stream>>>(Wv, WvT, D, D);
    hidden_t_kernel<<<4096, 256, 0, stream>>>(hidden, hT);

    // per-token gate scalars + per-batch decay scan
    token_scalars_kernel<<<1024, 256, 0, stream>>>(x, flw, flb, fww, fwb, zlr, zwd);
    scan_kernel<<<B, 256, 0, stream>>>(zlr, zwd, lbl, lbw, lr, C, wdc, coef, wdlast);

    // Q (transposed bf16), K (bf16), V (f32) — TDM-staged B panel
    gemm_xw_kernel<<<2048, 256, 0, stream>>>(Xbf, WqT, nullptr, nullptr, Qt, 0);
    gemm_xw_kernel<<<2048, 256, 0, stream>>>(Xbf, WkT, nullptr, Kbf, nullptr, 1);
    gemm_xw_kernel<<<2048, 256, 0, stream>>>(Xbf, WvT, V, nullptr, nullptr, 2);

    // WpK = K * W_prev^T : emits y_cross, WpK-V, Ut
    gemm_wpk_kernel<<<2048, 256, 0, stream>>>(Kbf, hT, V, wdc, coef, y, WpKV, Ut);

    // masked score row-sums with decay early-exit
    kq_rowsum_kernel<<<64, 256, 0, stream>>>(Kbf, Qt, C, S);

    // y = y_cross - WpKV * lr * S
    finalize_y_kernel<<<4096, 256, 0, stream>>>(y, WpKV, lr, S);

    // W_next = Ut * K + W_prev * wd_cross[-1]
    gemm_wnext_kernel<<<512, 256, 0, stream>>>(Ut, Kbf, hidden, wdlast, Wn);
}
